// TopoSTGformer_1357209666528
// MI455X (gfx1250) — matrix operand order repaired and checked by reading, compiled, not run on hardware
//
#include <hip/hip_runtime.h>
#include <hip/hip_bf16.h>

// ---------------------------------------------------------------------------
// TopoSTGformer forward for MI455X (gfx1250).
// All GEMM-shaped work runs through v_wmma_f32_16x16x32_f16 (f16 in, f32 acc).
// Problem is HBM-bandwidth bound (~115 GFLOP vs ~2-3 GB traffic @ 23.3 TB/s),
// so GEMMs stage the K x N operand once in LDS per workgroup and stream the
// row operand with branch-free float4 loads + inline cvt to f16.
// ---------------------------------------------------------------------------

typedef __attribute__((ext_vector_type(16))) _Float16 v16h;
typedef __attribute__((ext_vector_type(8)))  _Float16 v8h;
typedef __attribute__((ext_vector_type(8)))  float    v8f;
typedef __attribute__((ext_vector_type(4)))  float    v4f;

static constexpr int BB  = 64;
static constexpr int TT  = 12;
static constexpr int NN  = 716;
static constexpr int CCc = 60;
static constexpr long RR  = (long)BB * TT * NN;      // 549888 rows
static constexpr long R60 = RR * 60;                 // 32,993,280 floats

// ---------------------------------------------------------------------------
// Generic WMMA GEMM:  Out[b] = (A[b] (MxK, f32) * Bw[b] (KxNC, f32) + bias)*scale
// - Bw is converted to f16 and staged transposed in LDS: Blds[col][k].
// - 256 threads = 8 wave32; each wave owns one 16-row M tile, NT 16-col tiles.
// - a_mod != 0: A batch index = (batch % a_mod)  (graph[t] reuse over b*T+t).
// - Out-of-range M rows are CLAMPED for the A stream (no EXEC divergence);
//   their garbage D rows are masked at the store. B is zero-padded in K, so
//   the only k-guards live in a single tail iteration.
// ---------------------------------------------------------------------------
template <int NT, bool RELU>
__global__ __launch_bounds__(256) void gemm_f16w(
    const float* __restrict__ A, long a_bstride, int a_mod,
    const float* __restrict__ Bw, long b_bstride,
    const float* __restrict__ bias,
    float* __restrict__ Out, long o_bstride,
    int M, int K, int NC, int lda, int ldo, float scale)
{
  extern __shared__ _Float16 Blds[];
  const int KP  = (K + 31) & ~31;       // K padded to multiple of 32 (zeros)
  const int SP  = KP + 8;               // pad to de-phase LDS banks
  const int NCP = NT * 16;
  // Largest k0-prefix where the 32-wide A slab [k0, k0+31] is fully in range.
  const int KF  = (K >= 32) ? (((K - 32) / 32) * 32 + 32) : 0;

  const int batch = blockIdx.y;
  const float* Ab = A + (long)(a_mod ? (batch % a_mod) : batch) * a_bstride;
  const float* Bb = Bw + (long)batch * b_bstride;
  float*       Ob = Out + (long)batch * o_bstride;

  // Stage B: global row-major [k][col] -> LDS f16 transposed [col][k], padded.
  for (int i = threadIdx.x; i < NCP * KP; i += 256) {
    int col = i % NCP, k = i / NCP;
    float v = (k < K && col < NC) ? Bb[(long)k * NC + col] : 0.0f;
    Blds[col * SP + k] = (_Float16)v;
  }
  __syncthreads();

  const int lane = threadIdx.x & 31;
  const int wave = threadIdx.x >> 5;
  const int l16  = lane & 15;
  const int hi   = lane >> 4;           // 0 for lanes 0-15, 1 for 16-31
  const int tiles = (M + 15) >> 4;

  for (int tile = blockIdx.x * 8 + wave; tile < tiles; tile += gridDim.x * 8) {
    v8f acc[NT];
#pragma unroll
    for (int j = 0; j < NT; ++j)
#pragma unroll
      for (int r = 0; r < 8; ++r) acc[j][r] = 0.0f;

    // A lane row (lanes l and l+16 share row M=l); clamp instead of guard.
    int arow = tile * 16 + l16;
    arow = arow < (M - 1) ? arow : (M - 1);
    const float* Arow = Ab + (long)arow * lda;

    // B fragments for one k-step + the NT WMMAs (loads grouped before MMAs).
    auto mm_step = [&](const v16h& a, int k0) {
      v16h bfr[NT];
#pragma unroll
      for (int j = 0; j < NT; ++j) {
        // B fragment (32x16 f16): lane -> col = j*16+l16; K = 8*hi + e (e<8), 16 + 8*hi + (e-8) (e>=8).
        const _Float16* bp = &Blds[(j * 16 + l16) * SP + k0 + hi * 8];
        v8h blo = *(const v8h*)bp;
        v8h bhh = *(const v8h*)(bp + 16);
#pragma unroll
        for (int e = 0; e < 8; ++e) { bfr[j][e] = blo[e]; bfr[j][8 + e] = bhh[e]; }
      }
#pragma unroll
      for (int j = 0; j < NT; ++j) {
        acc[j] = __builtin_amdgcn_wmma_f32_16x16x32_f16(
            false, a, false, bfr[j], (short)0, acc[j], false, false);
      }
    };

    int k0 = 0;
    // Main body: branch-free vector loads.
    for (; k0 < KF; k0 += 32) {
      const int b0 = k0 + hi * 8;
      v4f p0 = *(const v4f*)(Arow + b0);
      v4f p1 = *(const v4f*)(Arow + b0 + 4);
      v4f p2 = *(const v4f*)(Arow + b0 + 16);
      v4f p3 = *(const v4f*)(Arow + b0 + 20);
      v16h a;
#pragma unroll
      for (int e = 0; e < 4; ++e) {
        a[e]      = (_Float16)p0[e];
        a[4 + e]  = (_Float16)p1[e];
        a[8 + e]  = (_Float16)p2[e];
        a[12 + e] = (_Float16)p3[e];
      }
      mm_step(a, k0);
    }
    // Tail (at most one iteration): k-guarded scalar loads.
    for (; k0 < KP; k0 += 32) {
      const int b0 = k0 + hi * 8;
      v16h a;
#pragma unroll
      for (int e = 0; e < 8; ++e) {
        int k1 = b0 + e, k2 = b0 + 16 + e;
        a[e]     = (k1 < K) ? (_Float16)Arow[k1] : (_Float16)0.0f;
        a[8 + e] = (k2 < K) ? (_Float16)Arow[k2] : (_Float16)0.0f;
      }
      mm_step(a, k0);
    }

    // Epilogue: C/D layout lane&15=N, VGPR r -> M = r + 8*hi.
#pragma unroll
    for (int j = 0; j < NT; ++j) {
      const int col = j * 16 + l16;
      if (col < NC) {
        float bv = bias ? bias[col] : 0.0f;
#pragma unroll
        for (int r = 0; r < 8; ++r) {
          int row = tile * 16 + r + hi * 8;
          if (row < M) {
            float v = (acc[j][r] + bv) * scale;
            if (RELU) v = v > 0.0f ? v : 0.0f;
            Ob[(long)row * ldo + col] = v;
          }
        }
      }
    }
  }
}

// ------------------------- small VALU kernels ------------------------------

__global__ void topo_kernel(const float* __restrict__ tf,
                            const float* __restrict__ w1, const float* __restrict__ b1,
                            const float* __restrict__ w2, const float* __restrict__ b2,
                            float* __restrict__ ztopo) {
  int n = blockIdx.x * blockDim.x + threadIdx.x;
  if (n >= NN) return;
  float z[8]; float nrm = 0.0f;
#pragma unroll
  for (int i = 0; i < 8; ++i) { float t = tf[n * 8 + i]; nrm += t * t; }
  nrm = sqrtf(nrm) + 1e-6f;
#pragma unroll
  for (int i = 0; i < 8; ++i) z[i] = log1pf(tf[n * 8 + i] / nrm);
  float out[8];
#pragma unroll
  for (int o = 0; o < 8; ++o) out[o] = b2[o];
  for (int j = 0; j < 64; ++j) {
    float h = b1[j];
#pragma unroll
    for (int i = 0; i < 8; ++i) h += z[i] * w1[i * 64 + j];
    h = h > 0.0f ? h : 0.0f;
#pragma unroll
    for (int o = 0; o < 8; ++o) out[o] += h * w2[j * 8 + o];
  }
#pragma unroll
  for (int o = 0; o < 8; ++o) ztopo[n * 8 + o] = out[o];
}

__global__ void qkadd_kernel(const float* __restrict__ ztopo,
                             const float* __restrict__ W, const float* __restrict__ bvec,
                             const float* __restrict__ beta, float* __restrict__ out) {
  int i = blockIdx.x * blockDim.x + threadIdx.x;  // n*60 + c
  if (i >= NN * CCc) return;
  int n = i / CCc, c = i % CCc;
  float s = bvec[c];
#pragma unroll
  for (int m = 0; m < 8; ++m) s += ztopo[n * 8 + m] * W[m * CCc + c];
  out[i] = beta[0] * s;
}

__global__ void embed_kernel(const float* __restrict__ x,
                             const float* __restrict__ inW, const float* __restrict__ inb,
                             const float* __restrict__ todtab, const float* __restrict__ dowtab,
                             const float* __restrict__ adp,
                             const int* __restrict__ tod, const int* __restrict__ dow,
                             float* __restrict__ h0) {
  long r = blockIdx.x * (long)blockDim.x + threadIdx.x;
  if (r >= RR) return;
  int n = (int)(r % NN);
  int t = (int)((r / NN) % TT);
  float x0 = x[r * 3], x1 = x[r * 3 + 1], x2 = x[r * 3 + 2];
  float* o = h0 + r * 60;
#pragma unroll
  for (int c = 0; c < 24; ++c) o[c] = x0 * inW[c] + x1 * inW[24 + c] + x2 * inW[48 + c] + inb[c];
  int td = tod[r], dw = dow[r];
#pragma unroll
  for (int c = 0; c < 12; ++c) o[24 + c] = todtab[td * 12 + c];
#pragma unroll
  for (int c = 0; c < 12; ++c) o[36 + c] = dowtab[dw * 12 + c];
#pragma unroll
  for (int c = 0; c < 12; ++c) o[48 + c] = adp[((long)t * NN + n) * 12 + c];
}

// q += qadd[n]; k += kadd[n]; then per-head L2 normalize q,k in place.
__global__ void qknorm_kernel(float* __restrict__ qkv,
                              const float* __restrict__ qadd, const float* __restrict__ kadd) {
  long i = blockIdx.x * (long)blockDim.x + threadIdx.x;  // r*4 + h
  if (i >= RR * 4) return;
  int h = (int)(i & 3);
  long r = i >> 2;
  int n = (int)(r % NN);
  float* q = qkv + r * 180 + h * 15;
  float* k = q + 60;
  const float* qa = qadd + (long)n * 60 + h * 15;
  const float* ka = kadd + (long)n * 60 + h * 15;
  float qv[15], kv[15], qn = 0.0f, kn = 0.0f;
#pragma unroll
  for (int m = 0; m < 15; ++m) {
    qv[m] = q[m] + qa[m]; qn += qv[m] * qv[m];
    kv[m] = k[m] + ka[m]; kn += kv[m] * kv[m];
  }
  qn = fmaxf(sqrtf(qn), 1e-12f);
  kn = fmaxf(sqrtf(kn), 1e-12f);
#pragma unroll
  for (int m = 0; m < 15; ++m) { q[m] = qv[m] / qn; k[m] = kv[m] / kn; }
}

// Spatial reduce per (b,t): kvs[h][m][d] = sum_n kn*v ; ks_sum[h][m] = sum_n kn.
__global__ __launch_bounds__(256) void sred_kernel(const float* __restrict__ qkv,
                                                   float* __restrict__ kvs) {
  __shared__ float kch[64 * 60];
  __shared__ float vch[64 * 60];
  int bt = blockIdx.x;
  const float* base = qkv + (long)bt * NN * 180;
  int tid = threadIdx.x;
  float accv[4] = {0.0f, 0.0f, 0.0f, 0.0f};
  for (int n0 = 0; n0 < NN; n0 += 64) {
    int cnt = (NN - n0) < 64 ? (NN - n0) : 64;
    __syncthreads();
    for (int i = tid; i < cnt * 60; i += 256) {
      int nl = i / 60, c = i % 60;
      kch[i] = base[(long)(n0 + nl) * 180 + 60 + c];
      vch[i] = base[(long)(n0 + nl) * 180 + 120 + c];
    }
    __syncthreads();
#pragma unroll
    for (int u = 0; u < 4; ++u) {
      int o = tid + u * 256;
      if (o < 960) {
        int h = o / 240, rem = o % 240;
        float s = accv[u];
        if (rem < 225) {
          int m = rem / 15, d = rem % 15;
          for (int nl = 0; nl < cnt; ++nl)
            s += kch[nl * 60 + h * 15 + m] * vch[nl * 60 + h * 15 + d];
        } else {
          int m = rem - 225;
          for (int nl = 0; nl < cnt; ++nl) s += kch[nl * 60 + h * 15 + m];
        }
        accv[u] = s;
      }
    }
  }
#pragma unroll
  for (int u = 0; u < 4; ++u) {
    int o = tid + u * 256;
    if (o < 960) kvs[(long)bt * 960 + o] = accv[u];
  }
}

// Spatial apply: out_s = (qn*kvs + L*v) / max(qn.ks_sum + L, 1e-5) -> attcat[:,0:60]
__global__ __launch_bounds__(256) void sapply_kernel(const float* __restrict__ qkv,
                                                     const float* __restrict__ kvs,
                                                     float* __restrict__ attcat) {
  __shared__ float kl[960];
  int bt = blockIdx.x;
  int tid = threadIdx.x;
  for (int i = tid; i < 960; i += 256) kl[i] = kvs[(long)bt * 960 + i];
  __syncthreads();
  int n = blockIdx.y * 64 + (tid >> 2);
  int h = tid & 3;
  if (n >= NN) return;
  long r = (long)bt * NN + n;
  const float* q  = qkv + r * 180 + h * 15;
  const float* v  = qkv + r * 180 + 120 + h * 15;
  const float* kv = kl + h * 240;       // [m*15+d]; ks_sum at 225+m
  float qv[15], den = 0.0f;
#pragma unroll
  for (int m = 0; m < 15; ++m) { qv[m] = q[m]; den += qv[m] * kv[225 + m]; }
  den = fmaxf(den + (float)NN, 1e-5f);
  float* o = attcat + r * 120 + h * 15;
#pragma unroll
  for (int d = 0; d < 15; ++d) {
    float num = (float)NN * v[d];
#pragma unroll
    for (int m = 0; m < 15; ++m) num += qv[m] * kv[m * 15 + d];
    o[d] = num / den;
  }
}

// Temporal attention (L = T = 12) -> attcat[:,60:120]
__global__ void tattn_kernel(const float* __restrict__ qkv, float* __restrict__ attcat) {
  long i = blockIdx.x * (long)blockDim.x + threadIdx.x;  // (b*N+n)*60 + h*15 + d
  if (i >= (long)BB * NN * 60) return;
  int d = (int)(i % 15);
  int h = (int)((i / 15) & 3);
  long bn = i / 60;
  int n = (int)(bn % NN);
  int b = (int)(bn / NN);
  float kvc[15], kss[15];
#pragma unroll
  for (int m = 0; m < 15; ++m) { kvc[m] = 0.0f; kss[m] = 0.0f; }
  for (int t = 0; t < TT; ++t) {
    long r = (long)(b * TT + t) * NN + n;
    const float* k = qkv + r * 180 + 60 + h * 15;
    float vd = qkv[r * 180 + 120 + h * 15 + d];
#pragma unroll
    for (int m = 0; m < 15; ++m) { kvc[m] += k[m] * vd; kss[m] += k[m]; }
  }
  for (int t = 0; t < TT; ++t) {
    long r = (long)(b * TT + t) * NN + n;
    const float* q = qkv + r * 180 + h * 15;
    float vd = qkv[r * 180 + 120 + h * 15 + d];
    float num = (float)TT * vd, den = (float)TT;
#pragma unroll
    for (int m = 0; m < 15; ++m) { num += q[m] * kvc[m]; den += q[m] * kss[m]; }
    attcat[r * 120 + 60 + h * 15 + d] = num / fmaxf(den, 1e-5f);
  }
}

// x_glo = (init ? h0 : x_glo) + att * gate
__global__ void glo_kernel(const float* __restrict__ att, const float* __restrict__ gate,
                           const float* __restrict__ h0, float* __restrict__ xglo, int init) {
  long i = blockIdx.x * (long)blockDim.x + threadIdx.x;
  if (i >= R60) return;
  float base = init ? h0[i] : xglo[i];
  xglo[i] = base + att[i] * gate[i];
}

// out = LayerNorm(a + b) * g + bi   (per row of 60, eps=1e-5)
__global__ void ln_kernel(const float* __restrict__ a, const float* __restrict__ b,
                          const float* __restrict__ g, const float* __restrict__ bi,
                          float* __restrict__ out) {
  long r = blockIdx.x * (long)blockDim.x + threadIdx.x;
  if (r >= RR) return;
  const float* pa = a + r * 60;
  const float* pb = b + r * 60;
  float buf[60], s = 0.0f;
#pragma unroll
  for (int c = 0; c < 60; ++c) { buf[c] = pa[c] + pb[c]; s += buf[c]; }
  float m = s / 60.0f, var = 0.0f;
#pragma unroll
  for (int c = 0; c < 60; ++c) { float d = buf[c] - m; var += d * d; }
  float inv = rsqrtf(var / 60.0f + 1e-5f);
  float* o = out + r * 60;
#pragma unroll
  for (int c = 0; c < 60; ++c) o[c] = (buf[c] - m) * inv * g[c] + bi[c];
}

// (b,t,n,c) -> (b, n, t*60+c)
__global__ void permute_kernel(const float* __restrict__ h2, float* __restrict__ perm) {
  long i = blockIdx.x * (long)blockDim.x + threadIdx.x;
  if (i >= (long)BB * NN * 720) return;
  int c = (int)(i % 60);
  int t = (int)((i / 60) % TT);
  int n = (int)((i / 720) % NN);
  long b = i / ((long)720 * NN);
  perm[i] = h2[((b * TT + t) * NN + n) * 60 + c];
}

// ---------------------------------------------------------------------------
static inline long cdivl(long a, long b) { return (a + b - 1) / b; }
static inline size_t ldsB(int K, int NT) {
  int KP = (K + 31) & ~31;
  return (size_t)(NT * 16) * (size_t)(KP + 8) * 2;
}

extern "C" void kernel_launch(void* const* d_in, const int* in_sizes, int n_in,
                              void* d_out, int out_size, void* d_ws, size_t ws_size,
                              hipStream_t stream) {
  (void)in_sizes; (void)n_in; (void)out_size; (void)ws_size;
  // Inputs flattened in setup_inputs() insertion order (DFS through params).
  const float* x       = (const float*)d_in[0];
  const float* graph   = (const float*)d_in[1];
  const float* topo    = (const float*)d_in[2];
  const float* in_W    = (const float*)d_in[3];
  const float* in_b    = (const float*)d_in[4];
  const float* tod_tab = (const float*)d_in[5];
  const float* dow_tab = (const float*)d_in[6];
  const float* adp     = (const float*)d_in[7];
  const float* te1_W   = (const float*)d_in[8];
  const float* te1_b   = (const float*)d_in[9];
  const float* te2_W   = (const float*)d_in[10];
  const float* te2_b   = (const float*)d_in[11];
  const float* enc_W   = (const float*)d_in[12];
  const float* enc_b   = (const float*)d_in[13];
  const float* qkvW[2] = {(const float*)d_in[14], (const float*)d_in[23]};
  const float* qkvB[2] = {(const float*)d_in[15], (const float*)d_in[24]};
  const float* qW[2]   = {(const float*)d_in[16], (const float*)d_in[25]};
  const float* qb[2]   = {(const float*)d_in[17], (const float*)d_in[26]};
  const float* kW[2]   = {(const float*)d_in[18], (const float*)d_in[27]};
  const float* kb[2]   = {(const float*)d_in[19], (const float*)d_in[28]};
  const float* beta[2] = {(const float*)d_in[20], (const float*)d_in[29]};
  const float* outW[2] = {(const float*)d_in[21], (const float*)d_in[30]};
  const float* outB[2] = {(const float*)d_in[22], (const float*)d_in[31]};
  const float* pwW[2]  = {(const float*)d_in[32], (const float*)d_in[33]};
  const float* pwB[2]  = {(const float*)d_in[34], (const float*)d_in[35]};
  const float* ln1g    = (const float*)d_in[36];
  const float* ln1b    = (const float*)d_in[37];
  const float* ln2g    = (const float*)d_in[38];
  const float* ln2b    = (const float*)d_in[39];
  const float* fc1W    = (const float*)d_in[40];
  const float* fc1b    = (const float*)d_in[41];
  const float* fc2W    = (const float*)d_in[42];
  const float* fc2b    = (const float*)d_in[43];
  const int*   tod     = (const int*)d_in[44];
  const int*   dow     = (const int*)d_in[45];
  float* out = (float*)d_out;

  // Workspace layout (floats). Needs ~1.46 GB.
  float* w     = (float*)d_ws;
  float* h0    = w;                 // embedding / residual / c0 / x_loc[0]
  float* xg    = w + 1 * R60;       // graph @ h0 (x_loc[1])
  float* qkvb  = w + 2 * R60;       // [3*R60] qkv; later reused as ffn mid
  float* attc  = w + 5 * R60;       // [2*R60] concat(out_s,out_t); later perm
  float* att0  = w + 7 * R60;       // iter0 attention out; later h2
  float* att1  = w + 8 * R60;       // iter1 attention out; later x (post-LN1)
  float* gateb = w + 9 * R60;       // gate; later fc2 out
  float* xglo  = w + 10 * R60;
  float* ztopo = w + 11 * R60;      // 5728
  float* qkadd = ztopo + 8192;      // 4 * N*60 buffers: (o*2+s)
  float* kvsb  = qkadd + 4L * NN * 60;  // (B*T)*960

  // 1) topo embedding + topo q/k biases (beta folded in)
  topo_kernel<<<(int)cdivl(NN, 64), 64, 0, stream>>>(topo, te1_W, te1_b, te2_W, te2_b, ztopo);
  for (int o = 0; o < 2; ++o) {
    qkadd_kernel<<<(int)cdivl((long)NN * 60, 256), 256, 0, stream>>>(
        ztopo, qW[o], qb[o], beta[o], qkadd + (long)(o * 2 + 0) * NN * 60);
    qkadd_kernel<<<(int)cdivl((long)NN * 60, 256), 256, 0, stream>>>(
        ztopo, kW[o], kb[o], beta[o], qkadd + (long)(o * 2 + 1) * NN * 60);
  }

  // 2) input embedding concat -> h0 (B,T,N,60)
  embed_kernel<<<(int)cdivl(RR, 256), 256, 0, stream>>>(
      x, in_W, in_b, tod_tab, dow_tab, adp, tod, dow, h0);

  // 3) graph propagation: xg[b,t] = graph[t] @ h0[b,t]   (716x716x60, WMMA)
  {
    int tiles = (int)cdivl(NN, 16);  // 45
    gemm_f16w<4, false><<<dim3((unsigned)cdivl(tiles, 8), BB * TT), 256,
                          ldsB(NN, 4), stream>>>(
        graph, (long)NN * NN, TT, h0, (long)NN * 60, nullptr,
        xg, (long)NN * 60, NN, NN, 60, NN, 60, 1.0f);
  }

  const int tilesR = (int)cdivl(RR, 16);        // 34368
  const unsigned gR = (unsigned)cdivl(tilesR, 8);

  // 4) two attention orders
  for (int o = 0; o < 2; ++o) {
    const float* xin = o ? xg : h0;
    // qkv projection (60 -> 180), WMMA
    gemm_f16w<12, false><<<dim3(gR, 1), 256, ldsB(60, 12), stream>>>(
        xin, 0, 0, qkvW[o], 0, qkvB[o], qkvb, 0, (int)RR, 60, 180, 60, 180, 1.0f);
    // topo bias + per-head normalization (in place)
    qknorm_kernel<<<(int)cdivl(RR * 4, 256), 256, 0, stream>>>(
        qkvb, qkadd + (long)(o * 2 + 0) * NN * 60, qkadd + (long)(o * 2 + 1) * NN * 60);
    // spatial linear attention
    sred_kernel<<<BB * TT, 256, 0, stream>>>(qkvb, kvsb);
    sapply_kernel<<<dim3(BB * TT, (unsigned)cdivl(NN, 64)), 256, 0, stream>>>(qkvb, kvsb, attc);
    // temporal linear attention
    tattn_kernel<<<(int)cdivl((long)BB * NN * 60, 256), 256, 0, stream>>>(qkvb, attc);
    // output projection (120 -> 60), WMMA
    float* atv = o ? att1 : att0;
    gemm_f16w<4, false><<<dim3(gR, 1), 256, ldsB(120, 4), stream>>>(
        attc, 0, 0, outW[o], 0, outB[o], atv, 0, (int)RR, 120, 60, 120, 60, 1.0f);
    // gate = (c @ pw_W + pw_b) * SCALE[o], WMMA
    const float* cbuf = o ? att0 : h0;
    gemm_f16w<4, false><<<dim3(gR, 1), 256, ldsB(60, 4), stream>>>(
        cbuf, 0, 0, pwW[o], 0, pwB[o], gateb, 0, (int)RR, 60, 60, 60, 60,
        o ? 0.01f : 1.0f);
    // x_glo accumulate (seeded from h0 on first iteration)
    glo_kernel<<<(int)cdivl(R60, 256), 256, 0, stream>>>(atv, gateb, h0, xglo, o == 0 ? 1 : 0);
  }

  // 5) LN1, FFN (WMMA), LN2
  float* xb   = att1;   // att1 consumed
  float* ffnm = qkvb;   // qkv consumed
  float* fc2o = gateb;  // gate consumed
  float* h2   = att0;   // att0 consumed
  ln_kernel<<<(int)cdivl(RR, 256), 256, 0, stream>>>(h0, xglo, ln1g, ln1b, xb);
  gemm_f16w<8, true><<<dim3(gR, 1), 256, ldsB(60, 8), stream>>>(
      xb, 0, 0, fc1W, 0, fc1b, ffnm, 0, (int)RR, 60, 120, 60, 120, 1.0f);
  gemm_f16w<4, false><<<dim3(gR, 1), 256, ldsB(120, 4), stream>>>(
      ffnm, 0, 0, fc2W, 0, fc2b, fc2o, 0, (int)RR, 120, 60, 120, 60, 1.0f);
  ln_kernel<<<(int)cdivl(RR, 256), 256, 0, stream>>>(xb, fc2o, ln2g, ln2b, h2);

  // 6) (b,t,n,c) -> (b,n,720), then final encoder GEMM (720 -> 60) into d_out
  float* permb = attc;
  permute_kernel<<<(int)cdivl((long)BB * NN * 720, 256), 256, 0, stream>>>(h2, permb);
  {
    int Me = BB * NN;  // 45824
    gemm_f16w<4, false><<<dim3((unsigned)cdivl(cdivl(Me, 16), 8), 1), 256,
                          ldsB(720, 4), stream>>>(
        permb, 0, 0, enc_W, 0, enc_b, out, 0, Me, 720, 60, 720, 60, 1.0f);
  }
}
